// TransformerBlock_33182917329237
// MI455X (gfx1250) — compile-verified, matrix operand order checked by reading, not run on hardware
//
#include <hip/hip_runtime.h>
#include <hip/hip_bf16.h>
#include <math.h>

#define B_   2
#define S_   2048
#define D_   1536
#define H_   16
#define KV_  4
#define HD_  96
#define NREP_ 4
#define DFF_ 4096
#define DSH_ 2048
#define E_   8
#define NTOK (B_ * S_)

typedef __bf16 v16bf __attribute__((ext_vector_type(16)));
typedef float  v8f   __attribute__((ext_vector_type(8)));

union BFrag {
    unsigned short s[16];
    unsigned int   u[8];
    uint4          u4[2];
    v16bf          v;
};

// scalar fallback converter (RNE) for scattered single stores
__device__ __forceinline__ unsigned short f2bf(float f) {
    unsigned u = __builtin_bit_cast(unsigned, f);
    u += 0x7FFFu + ((u >> 16) & 1u);
    return (unsigned short)(u >> 16);
}

// packed 2 x f32 -> 2 x bf16 in one dword (lowers to v_cvt_pk_bf16_f32 when available)
__device__ __forceinline__ unsigned pack_bf16x2(float a, float b) {
    __hip_bfloat162 h2 = __float22bfloat162_rn(make_float2(a, b));
    unsigned r;
    __builtin_memcpy(&r, &h2, sizeof(r));
    return r;
}

__device__ __forceinline__ v8f wmma_bf16(v16bf a, v16bf b, v8f c) {
    // D = A(16x32 bf16) * B(32x16 bf16) + C(16x16 f32)
    return __builtin_amdgcn_wmma_f32_16x16x32_bf16(false, a, false, b, (short)0, c,
                                                   false, false);
}

// ---------------------------------------------------------------------------
// Tiled GEMM:  C[N x M] = A[N x K] * W[M x K]^T  (+ add_src, * row_scale, += C)
// block = 256 threads (8 waves), 128x128 tile, K-step 32, bf16 WMMA, f32 accum
// Requires N, M multiples of 128 and K multiple of 32.
// ---------------------------------------------------------------------------
__global__ __launch_bounds__(256)
void gemm_bf16_wmma_kernel(const float* __restrict__ A, const float* __restrict__ W,
                           float* __restrict__ C, int N, int K, int M,
                           const float* __restrict__ add_src,
                           const float* __restrict__ row_scale, int rs_stride,
                           int accumulate)
{
    constexpr int LSTR = 40;                    // bf16 elems/row: 20 dwords, bank-conflict free
    __shared__ unsigned short lA[128 * LSTR];
    __shared__ unsigned short lB[128 * LSTR];

    const int tid  = threadIdx.x;
    const int lane = tid & 31, wid = tid >> 5;
    const int wrow = wid & 3,  wcol = wid >> 2;         // waves: 4 (rows) x 2 (cols)
    const int rowBase = blockIdx.y * 128;               // token rows
    const int colBase = blockIdx.x * 128;               // feature cols
    const int hi = lane >> 4, ln = lane & 15;

    v8f acc[2][4];
    const v8f vzero = {0.f, 0.f, 0.f, 0.f, 0.f, 0.f, 0.f, 0.f};
#pragma unroll
    for (int i = 0; i < 2; ++i)
#pragma unroll
        for (int j = 0; j < 4; ++j) acc[i][j] = vzero;

    // staging map: 2 threads per row, 16 f32 each
    const int lr = tid >> 1;
    const int lc = (tid & 1) * 16;
    const float* aSrc = A + (size_t)(rowBase + lr) * K + lc;
    const float* bSrc = W + (size_t)(colBase + lr) * K + lc;
    unsigned short* aDst = &lA[lr * LSTR + lc];
    unsigned short* bDst = &lB[lr * LSTR + lc];

    for (int k0 = 0; k0 < K; k0 += 32) {
        __syncthreads();
#pragma unroll
        for (int j = 0; j < 4; ++j) {
            float4 fa = *(const float4*)(aSrc + k0 + j * 4);
            float4 fb = *(const float4*)(bSrc + k0 + j * 4);
            *(unsigned int*)&aDst[j * 4]     = pack_bf16x2(fa.x, fa.y);
            *(unsigned int*)&aDst[j * 4 + 2] = pack_bf16x2(fa.z, fa.w);
            *(unsigned int*)&bDst[j * 4]     = pack_bf16x2(fb.x, fb.y);
            *(unsigned int*)&bDst[j * 4 + 2] = pack_bf16x2(fb.z, fb.w);
        }
        if (k0 + 32 < K) {                     // global_prefetch_b8 for next K tile
            __builtin_prefetch(aSrc + k0 + 32, 0, 0);
            __builtin_prefetch(bSrc + k0 + 32, 0, 0);
        }
        __syncthreads();

        BFrag af[2], bf[4];
        const int koff = hi * 8;               // ISA 16-bit A layout: hi half holds K+8 / K+24
#pragma unroll
        for (int fr = 0; fr < 2; ++fr) {
            const int r = wrow * 32 + fr * 16 + ln;
            af[fr].u4[0] = *(const uint4*)&lA[r * LSTR + koff];
            af[fr].u4[1] = *(const uint4*)&lA[r * LSTR + koff + 16];
        }
#pragma unroll
        for (int fc = 0; fc < 4; ++fc) {
            const int cc = wcol * 64 + fc * 16 + ln;
            bf[fc].u4[0] = *(const uint4*)&lB[cc * LSTR + koff];
            bf[fc].u4[1] = *(const uint4*)&lB[cc * LSTR + koff + 16];
        }
#pragma unroll
        for (int fr = 0; fr < 2; ++fr)
#pragma unroll
            for (int fc = 0; fc < 4; ++fc)
                acc[fr][fc] = wmma_bf16(af[fr].v, bf[fc].v, acc[fr][fc]);
    }

    // epilogue: C layout lane = (n = ln, m = r + 8*hi)
#pragma unroll
    for (int fr = 0; fr < 2; ++fr)
#pragma unroll
        for (int fc = 0; fc < 4; ++fc)
#pragma unroll
            for (int r = 0; r < 8; ++r) {
                const int gr = rowBase + wrow * 32 + fr * 16 + r + 8 * hi;
                const int gc = colBase + wcol * 64 + fc * 16 + ln;
                float val = acc[fr][fc][r];
                if (row_scale) val *= row_scale[(size_t)gr * rs_stride];
                if (add_src)   val += add_src[(size_t)gr * M + gc];
                const size_t idx = (size_t)gr * M + gc;
                if (accumulate) val += C[idx];
                C[idx] = val;
            }
}

// ---------------------------------------------------------------------------
// Row-wise RMSNorm (D elems per row)
// ---------------------------------------------------------------------------
__global__ __launch_bounds__(256)
void rmsnorm_kernel(const float* __restrict__ x, const float* __restrict__ w,
                    float* __restrict__ out, int D)
{
    const int row = blockIdx.x;
    const float* xr = x + (size_t)row * D;
    float ss = 0.f;
    for (int i = threadIdx.x; i < D; i += 256) { const float v = xr[i]; ss += v * v; }
#pragma unroll
    for (int off = 16; off > 0; off >>= 1) ss += __shfl_xor(ss, off);
    __shared__ float red[8];
    const int lane = threadIdx.x & 31, wid = threadIdx.x >> 5;
    if (lane == 0) red[wid] = ss;
    __syncthreads();
    float tot = 0.f;
#pragma unroll
    for (int i = 0; i < 8; ++i) tot += red[i];
    const float inv = rsqrtf(tot / (float)D + 1e-6f);
    for (int i = threadIdx.x; i < D; i += 256)
        out[(size_t)row * D + i] = xr[i] * inv * w[i];
}

// ---------------------------------------------------------------------------
// Per-head RMSNorm + RoPE, in place.  grid = (token, head 0..19), block = 128
// ---------------------------------------------------------------------------
__global__ __launch_bounds__(128)
void qknorm_rope_kernel(float* __restrict__ q, float* __restrict__ k,
                        const float* __restrict__ qn_w, const float* __restrict__ kn_w)
{
    const int n = blockIdx.x;
    const int head = blockIdx.y;
    const int i = threadIdx.x;
    const int s = n & (S_ - 1);
    float* ptr; const float* wv;
    if (head < H_) { ptr = q + (size_t)n * (H_ * HD_) + head * HD_;        wv = qn_w; }
    else           { ptr = k + (size_t)n * (KV_ * HD_) + (head - H_) * HD_; wv = kn_w; }

    __shared__ float sv[128];
    __shared__ float sn[96];
    const float v = (i < HD_) ? ptr[i] : 0.f;
    sv[i] = v * v;
    __syncthreads();
#pragma unroll
    for (int off = 64; off > 0; off >>= 1) {
        if (i < off) sv[i] += sv[i + off];
        __syncthreads();
    }
    const float inv = rsqrtf(sv[0] / (float)HD_ + 1e-6f);
    if (i < HD_) sn[i] = v * inv * wv[i];
    __syncthreads();
    if (i < HD_) {
        const float rot = (i < 48) ? -sn[i + 48] : sn[i - 48];
        const float ang = (float)s * __powf(10000.f, -(float)(i % 48) / 48.f);
        ptr[i] = sn[i] * __cosf(ang) + rot * __sinf(ang);
    }
}

// ---------------------------------------------------------------------------
// Flash attention: grid = (S/128, H, B), block = 256 (8 waves x 16 query rows)
// q: [N, H*HD], k/v: [N, KV*HD], ao: [N, H*HD]. Causal, online softmax.
// ---------------------------------------------------------------------------
__global__ __launch_bounds__(256)
void attention_kernel(const float* __restrict__ q, const float* __restrict__ k,
                      const float* __restrict__ v, float* __restrict__ ao)
{
    const int qblk = blockIdx.x, h = blockIdx.y, b = blockIdx.z;
    const int kh = h / NREP_;
    const int tid = threadIdx.x, lane = tid & 31, wid = tid >> 5;
    const int hi = lane >> 4, ln = lane & 15;
    const int qbase = qblk * 128 + wid * 16;

    __shared__ unsigned short ldsP[8][16 * 32];  // per-wave P transpose buffer

    // Q fragments (A layout), 3 chunks of K-dim 32 covering HD=96
    BFrag qf[3];
    {
        const float* qrow = q + (size_t)(b * S_ + qbase + ln) * (H_ * HD_) + h * HD_ + hi * 8;
#pragma unroll
        for (int c = 0; c < 3; ++c) {
            const float* p = qrow + c * 32;
#pragma unroll
            for (int j = 0; j < 4; ++j) qf[c].u[j]     = pack_bf16x2(p[2 * j], p[2 * j + 1]);
#pragma unroll
            for (int j = 0; j < 4; ++j) qf[c].u[4 + j] = pack_bf16x2(p[16 + 2 * j], p[17 + 2 * j]);
        }
    }

    const v8f vzero = {0.f, 0.f, 0.f, 0.f, 0.f, 0.f, 0.f, 0.f};
    v8f o[6];
#pragma unroll
    for (int f = 0; f < 6; ++f) o[f] = vzero;
    float mrun[8], lrun[8];
#pragma unroll
    for (int r = 0; r < 8; ++r) { mrun[r] = -3.0e38f; lrun[r] = 0.f; }

    const float scale = 0.1020620726f;   // 1/sqrt(96)
    const float* kbp = k + (size_t)(b * S_) * (KV_ * HD_) + kh * HD_;
    const float* vbp = v + (size_t)(b * S_) * (KV_ * HD_) + kh * HD_;
    const int ntiles = (qbase + 15) / 32 + 1;   // causal: keys 0 .. qbase+15

    for (int t = 0; t < ntiles; ++t) {
        const int key0 = t * 32;
        v8f sf[2];
        sf[0] = vzero; sf[1] = vzero;
#pragma unroll
        for (int kc = 0; kc < 2; ++kc) {
            const float* kr = kbp + (size_t)(key0 + kc * 16 + ln) * (KV_ * HD_) + hi * 8;
#pragma unroll
            for (int c = 0; c < 3; ++c) {
                BFrag kf;
                const float* p = kr + c * 32;
#pragma unroll
                for (int j = 0; j < 4; ++j) kf.u[j]     = pack_bf16x2(p[2 * j], p[2 * j + 1]);
#pragma unroll
                for (int j = 0; j < 4; ++j) kf.u[4 + j] = pack_bf16x2(p[16 + 2 * j], p[17 + 2 * j]);
                sf[kc] = wmma_bf16(qf[c].v, kf.v, sf[kc]);
            }
        }
        // scale + causal mask + per-row tile max (16-lane xor reductions stay in half)
        float tmax[8];
#pragma unroll
        for (int r = 0; r < 8; ++r) {
            const int qr = qbase + r + 8 * hi;
            float s0 = sf[0][r] * scale;
            float s1 = sf[1][r] * scale;
            if (key0 + ln > qr)      s0 = -3.0e38f;
            if (key0 + 16 + ln > qr) s1 = -3.0e38f;
            sf[0][r] = s0; sf[1][r] = s1;
            float tm = fmaxf(s0, s1);
#pragma unroll
            for (int off = 1; off < 16; off <<= 1) tm = fmaxf(tm, __shfl_xor(tm, off));
            tmax[r] = tm;
        }
        float alpha[8];
#pragma unroll
        for (int r = 0; r < 8; ++r) {
            const float mn = fmaxf(mrun[r], tmax[r]);
            alpha[r] = __expf(mrun[r] - mn);
            mrun[r] = mn;
        }
#pragma unroll
        for (int r = 0; r < 8; ++r) {
            const float p0 = __expf(sf[0][r] - mrun[r]);
            const float p1 = __expf(sf[1][r] - mrun[r]);
            sf[0][r] = p0; sf[1][r] = p1;
            float rs = p0 + p1;
#pragma unroll
            for (int off = 1; off < 16; off <<= 1) rs += __shfl_xor(rs, off);
            lrun[r] = lrun[r] * alpha[r] + rs;
        }
#pragma unroll
        for (int f = 0; f < 6; ++f)
#pragma unroll
            for (int r = 0; r < 8; ++r) o[f][r] *= alpha[r];

        // P: C layout -> LDS -> A layout (same-wave LDS, in-order pipeline)
        unsigned short* pw = ldsP[wid];
#pragma unroll
        for (int r = 0; r < 8; ++r) {
            pw[(r + 8 * hi) * 32 + ln]      = f2bf(sf[0][r]);
            pw[(r + 8 * hi) * 32 + 16 + ln] = f2bf(sf[1][r]);
        }
        BFrag pf;
        pf.u4[0] = *(const uint4*)&pw[ln * 32 + hi * 8];
        pf.u4[1] = *(const uint4*)&pw[ln * 32 + hi * 8 + 16];

        // O += P(16x32) * V(32x96): 6 WMMAs, V loaded strided as B^T A-layout
#pragma unroll
        for (int f = 0; f < 6; ++f) {
            BFrag vf;
            const float* vp = vbp + (size_t)(key0 + hi * 8) * (KV_ * HD_) + f * 16 + ln;
#pragma unroll
            for (int j = 0; j < 4; ++j)
                vf.u[j] = pack_bf16x2(vp[(size_t)(2 * j) * (KV_ * HD_)],
                                      vp[(size_t)(2 * j + 1) * (KV_ * HD_)]);
#pragma unroll
            for (int j = 0; j < 4; ++j)
                vf.u[4 + j] = pack_bf16x2(vp[(size_t)(16 + 2 * j) * (KV_ * HD_)],
                                          vp[(size_t)(17 + 2 * j) * (KV_ * HD_)]);
            o[f] = wmma_bf16(pf.v, vf.v, o[f]);
        }
    }
#pragma unroll
    for (int f = 0; f < 6; ++f)
#pragma unroll
        for (int r = 0; r < 8; ++r) {
            const int qr = qbase + r + 8 * hi;
            ao[(size_t)(b * S_ + qr) * (H_ * HD_) + h * HD_ + f * 16 + ln] = o[f][r] / lrun[r];
        }
}

// ---------------------------------------------------------------------------
// Router: logits, softmax, top-2 combine weights, loss accumulators
// acc layout: [0]=sum lse^2, [1..8]=counts, [9..16]=sum probs
// ---------------------------------------------------------------------------
__global__ __launch_bounds__(256)
void gate_topk_kernel(const float* __restrict__ h2, const float* __restrict__ gate_w,
                      float* __restrict__ cw, float* __restrict__ acc)
{
    const int n = blockIdx.x;
    const float* xr = h2 + (size_t)n * D_;
    float part[E_];
#pragma unroll
    for (int e = 0; e < E_; ++e) part[e] = 0.f;
    for (int i = threadIdx.x; i < D_; i += 256) {
        const float xv = xr[i];
#pragma unroll
        for (int e = 0; e < E_; ++e) part[e] += xv * gate_w[e * D_ + i];
    }
#pragma unroll
    for (int e = 0; e < E_; ++e)
#pragma unroll
        for (int off = 16; off > 0; off >>= 1) part[e] += __shfl_xor(part[e], off);
    __shared__ float sh[8][E_];
    const int lane = threadIdx.x & 31, wid = threadIdx.x >> 5;
    if (lane == 0)
#pragma unroll
        for (int e = 0; e < E_; ++e) sh[wid][e] = part[e];
    __syncthreads();
    if (threadIdx.x == 0) {
        float lg[E_];
        for (int e = 0; e < E_; ++e) {
            float sum = 0.f;
            for (int w2 = 0; w2 < 8; ++w2) sum += sh[w2][e];
            lg[e] = sum;
        }
        float mx = lg[0];
        for (int e = 1; e < E_; ++e) mx = fmaxf(mx, lg[e]);
        float se = 0.f, w[E_];
        for (int e = 0; e < E_; ++e) { w[e] = __expf(lg[e] - mx); se += w[e]; }
        const float lse = mx + __logf(se);
        for (int e = 0; e < E_; ++e) w[e] /= se;
        int i1 = 0;
        for (int e = 1; e < E_; ++e) if (w[e] > w[i1]) i1 = e;
        int i2 = (i1 == 0) ? 1 : 0;
        for (int e = 0; e < E_; ++e) if (e != i1 && w[e] > w[i2]) i2 = e;
        const float tw = w[i1] + w[i2];
        cw[(size_t)n * E_ + i1] = w[i1] / tw;
        cw[(size_t)n * E_ + i2] = w[i2] / tw;
        atomicAdd(&acc[0], lse * lse);
        atomicAdd(&acc[1 + i1], 1.0f);
        atomicAdd(&acc[1 + i2], 1.0f);
        for (int e = 0; e < E_; ++e) atomicAdd(&acc[9 + e], w[e]);
    }
}

__global__ void swiglu_kernel(const float* __restrict__ g, const float* __restrict__ u,
                              float* __restrict__ t, size_t count)
{
    for (size_t i = blockIdx.x * 256ull + threadIdx.x; i < count;
         i += (size_t)gridDim.x * 256ull) {
        const float gv = g[i];
        t[i] = gv / (1.f + __expf(-gv)) * u[i];
    }
}

__global__ void finalize_kernel(const float* __restrict__ acc, float* __restrict__ tail)
{
    if (threadIdx.x == 0 && blockIdx.x == 0) {
        const float Nf = (float)NTOK;
        float bal = 0.f;
        for (int e = 0; e < E_; ++e)
            bal += (acc[9 + e] / Nf) * (acc[1 + e] / (Nf * 2.f));
        tail[0] = 0.001f * (acc[0] / Nf) + 0.01f * 8.f * bal;
        for (int e = 0; e < E_; ++e) tail[1 + e] = acc[1 + e];
    }
}

// ---------------------------------------------------------------------------
extern "C" void kernel_launch(void* const* d_in, const int* in_sizes, int n_in,
                              void* d_out, int out_size, void* d_ws, size_t ws_size,
                              hipStream_t stream)
{
    (void)in_sizes; (void)n_in; (void)out_size; (void)ws_size;
    const float* x      = (const float*)d_in[0];
    const float* n1w    = (const float*)d_in[1];
    const float* n2w    = (const float*)d_in[2];
    const float* q_w    = (const float*)d_in[3];
    const float* k_w    = (const float*)d_in[4];
    const float* v_w    = (const float*)d_in[5];
    const float* o_w    = (const float*)d_in[6];
    const float* qn_w   = (const float*)d_in[7];
    const float* kn_w   = (const float*)d_in[8];
    const float* gate_w = (const float*)d_in[9];
    const float* e_w1   = (const float*)d_in[10];
    const float* e_w2   = (const float*)d_in[11];
    const float* e_w3   = (const float*)d_in[12];
    const float* s_w1   = (const float*)d_in[13];
    const float* s_w2   = (const float*)d_in[14];
    const float* s_w3   = (const float*)d_in[15];
    float* out = (float*)d_out;

    float* ws = (float*)d_ws;
    size_t off = 0;
    float* h    = ws + off; off += (size_t)NTOK * D_;           // rmsnorm out (reused)
    float* qb   = ws + off; off += (size_t)NTOK * (H_ * HD_);
    float* kb   = ws + off; off += (size_t)NTOK * (KV_ * HD_);
    float* vb   = ws + off; off += (size_t)NTOK * (KV_ * HD_);
    float* ao   = ws + off; off += (size_t)NTOK * (H_ * HD_);
    float* x1   = ws + off; off += (size_t)NTOK * D_;
    float* cw   = ws + off; off += (size_t)NTOK * E_;
    float* gbuf = ws + off; off += (size_t)NTOK * DFF_;
    float* ubuf = ws + off; off += (size_t)NTOK * DFF_;
    float* tbuf = ws + off; off += (size_t)NTOK * DFF_;
    float* acc  = ws + off; off += 32;

    (void)hipMemsetAsync(cw, 0, (size_t)NTOK * E_ * sizeof(float), stream);
    (void)hipMemsetAsync(acc, 0, 32 * sizeof(float), stream);

    dim3 blk(256);
    auto gemm = [&](const float* A, const float* W, float* C, int n, int kk, int m,
                    const float* add, const float* rs, int rss, int accf) {
        dim3 grid(m / 128, n / 128);
        gemm_bf16_wmma_kernel<<<grid, blk, 0, stream>>>(A, W, C, n, kk, m, add, rs, rss, accf);
    };

    // ---- attention ----
    rmsnorm_kernel<<<NTOK, 256, 0, stream>>>(x, n1w, h, D_);
    gemm(h, q_w, qb, NTOK, D_, H_ * HD_, nullptr, nullptr, 0, 0);
    gemm(h, k_w, kb, NTOK, D_, KV_ * HD_, nullptr, nullptr, 0, 0);
    gemm(h, v_w, vb, NTOK, D_, KV_ * HD_, nullptr, nullptr, 0, 0);
    qknorm_rope_kernel<<<dim3(NTOK, H_ + KV_), 128, 0, stream>>>(qb, kb, qn_w, kn_w);
    attention_kernel<<<dim3(S_ / 128, H_, B_), 256, 0, stream>>>(qb, kb, vb, ao);
    gemm(ao, o_w, x1, NTOK, H_ * HD_, D_, x, nullptr, 0, 0);      // x1 = x + ao @ o_w.T

    // ---- MoE ----
    rmsnorm_kernel<<<NTOK, 256, 0, stream>>>(x1, n2w, h, D_);
    gate_topk_kernel<<<NTOK, 256, 0, stream>>>(h, gate_w, cw, acc);

    // shared SwiGLU
    gemm(h, s_w1, gbuf, NTOK, D_, DSH_, nullptr, nullptr, 0, 0);
    gemm(h, s_w3, ubuf, NTOK, D_, DSH_, nullptr, nullptr, 0, 0);
    swiglu_kernel<<<2048, 256, 0, stream>>>(gbuf, ubuf, tbuf, (size_t)NTOK * DSH_);
    gemm(tbuf, s_w2, out, NTOK, DSH_, D_, x1, nullptr, 0, 0);     // out = x1 + shared

    // routed experts (dense combine; zero weights contribute nothing)
    for (int e = 0; e < E_; ++e) {
        const float* w1 = e_w1 + (size_t)e * DFF_ * D_;
        const float* w2 = e_w2 + (size_t)e * D_ * DFF_;
        const float* w3 = e_w3 + (size_t)e * DFF_ * D_;
        gemm(h, w1, gbuf, NTOK, D_, DFF_, nullptr, nullptr, 0, 0);
        gemm(h, w3, ubuf, NTOK, D_, DFF_, nullptr, nullptr, 0, 0);
        swiglu_kernel<<<2048, 256, 0, stream>>>(gbuf, ubuf, tbuf, (size_t)NTOK * DFF_);
        gemm(tbuf, w2, out, NTOK, DFF_, D_, nullptr, cw + e, E_, 1); // out += cw[:,e]*(...)
    }

    finalize_kernel<<<1, 32, 0, stream>>>(acc, out + (size_t)NTOK * D_);
}
